// WaveNetDecoder_3126736191546
// MI455X (gfx1250) — matrix operand order repaired and checked by reading, not compiled, not run on hardware
//
#include <hip/hip_runtime.h>
#include <stdint.h>

#define TLEN   16384
#define WIDTH  512
#define SKIPW  256
#define NSTACK 30
#define NLAYERS 10
#define BOT    16
#define QUANTN 256
#define TEMB   32

typedef __attribute__((ext_vector_type(16))) __bf16 v16bf;
typedef __attribute__((ext_vector_type(8)))  float  v8f;

struct __align__(8) US4 { unsigned short x, y, z, w; };

__device__ __forceinline__ unsigned short f2bf(float f) {
    union { float f; uint32_t u; } c; c.f = f;
    uint32_t u = c.u;
    u += 0x7FFFu + ((u >> 16) & 1u);   // round-to-nearest-even
    return (unsigned short)(u >> 16);
}

// ---------------------------------------------------------------------------
// WMMA bf16 16x16x32 fragment-order indices (per CDNA5 ISA register layouts).
// A 16x32 (MxK): lanes 0-15 hold K 0..7 & 16..23, lanes 16-31 hold K 8..15 & 24..31
// ---------------------------------------------------------------------------
__device__ __forceinline__ int a_swz(int row, int k) {        // row 0..15, k 0..31
    int lane = (((k >> 3) & 1) << 4) | row;
    int e    = ((k >> 4) << 3) | (k & 7);
    return lane * 16 + e;
}
// B 32x16 (KxN): lanes 0-15 hold K 0..7 & 16..23 (N=lane), lanes 16-31 hold K 8..15 & 24..31
__device__ __forceinline__ int b_swz(int k, int n) {          // k 0..31, n 0..15
    int lane = (((k >> 3) & 1) << 4) | n;
    int e    = ((k >> 4) << 3) | (k & 7);
    return lane * 16 + e;
}

__device__ __forceinline__ v16bf frag_ld(const unsigned short* base) {
    return *(const v16bf*)(base + (threadIdx.x & 31) * 16);   // 32B contiguous per lane
}

// Stage a 32(K) x 64(N) B tile into fragment-order LDS.
// Quad-of-K mapping: 4 coalesced u16 global loads -> one contiguous b64 LDS store.
__device__ __forceinline__ void stage_B(const unsigned short* __restrict__ Bm,
                                        unsigned short* sB, int krow0, int t0, int shift) {
    const int tid = threadIdx.x;
#pragma unroll
    for (int it = 0; it < 2; ++it) {
        int q  = tid + it * 256;              // 512 quads total
        int n  = q & 63, kq = (q >> 6) << 2;  // kq in {0,4,...,28}
        int tt = t0 + n - shift;
        US4 v = {0, 0, 0, 0};
        if (tt >= 0) {
            const unsigned short* src = Bm + (size_t)(krow0 + kq) * TLEN + tt;
            v.x = src[0]; v.y = src[TLEN]; v.z = src[2 * TLEN]; v.w = src[3 * TLEN];
        }
        *(US4*)(sB + (n >> 4) * 512 + b_swz(kq, n & 15)) = v;  // e consecutive in K
    }
}

// ---------------------------------------------------------------------------
// Weight conversion: f32 [M][K] (arbitrary strides) -> bf16 fragment-block
// layout [M/16][K/32][512]. kbits = log2(K).
// ---------------------------------------------------------------------------
__global__ void k_cvt_swz(const float* __restrict__ src, unsigned short* __restrict__ dst,
                          int M, int kbits, int rs, int cs) {
    int i = blockIdx.x * blockDim.x + threadIdx.x;
    int K = 1 << kbits;
    if (i >= (M << kbits)) return;
    int m = i >> kbits, k = i & (K - 1);
    float v = src[(size_t)m * rs + (size_t)k * cs];
    dst[((size_t)(m >> 4) * (K >> 5) + (k >> 5)) * 512 + a_swz(m & 15, k & 31)] = f2bf(v);
}

__global__ void k_relu_to_bf16(const float* __restrict__ src, unsigned short* __restrict__ dst, int n) {
    int i = blockIdx.x * blockDim.x + threadIdx.x;
    if (i < n) { float v = src[i]; dst[i] = f2bf(v > 0.f ? v : 0.f); }
}

// cond_all[l][co][te] = w_cond[l,co,:]@emb[:,te] + b_cond[l,co] + b_dil[l,co]
__global__ void k_cond(const float* __restrict__ emb, const float* __restrict__ w_cond,
                       const float* __restrict__ b_cond, const float* __restrict__ b_dil,
                       float* __restrict__ cond_all) {
    int idx = blockIdx.x * blockDim.x + threadIdx.x;
    if (idx >= NSTACK * 1024) return;
    const float* w = w_cond + (size_t)idx * BOT;
    float bias = b_cond[idx] + b_dil[idx];
#pragma unroll 4
    for (int te = 0; te < TEMB; ++te) {
        float s = bias;
#pragma unroll
        for (int c = 0; c < BOT; ++c) s += w[c] * emb[c * TEMB + te];
        cond_all[(size_t)idx * TEMB + te] = s;
    }
}

// fcond[co][te] = w_fcond[co,:]@emb[:,te] + b_fcond[co]
__global__ void k_fcond(const float* __restrict__ emb, const float* __restrict__ w,
                        const float* __restrict__ b, float* __restrict__ out) {
    int co = blockIdx.x * blockDim.x + threadIdx.x;
    if (co >= SKIPW) return;
    for (int te = 0; te < TEMB; ++te) {
        float s = b[co];
#pragma unroll
        for (int c = 0; c < BOT; ++c) s += w[co * BOT + c] * emb[c * TEMB + te];
        out[co * TEMB + te] = s;
    }
}

// Initial causal conv: Cin=1, K=3, dilation 1. h[co][t] (f32 + bf16 copies)
__global__ void k_init(const float* __restrict__ x, const float* __restrict__ w,
                       const float* __restrict__ b, float* __restrict__ h,
                       unsigned short* __restrict__ h_bf) {
    int t = blockIdx.x * blockDim.x + threadIdx.x;
    if (t >= TLEN) return;
    float x0 = (t >= 2) ? x[t - 2] : 0.f;
    float x1 = (t >= 1) ? x[t - 1] : 0.f;
    float x2 = x[t];
    for (int co = 0; co < WIDTH; ++co) {
        float v = w[co * 3 + 0] * x0 + w[co * 3 + 1] * x1 + w[co * 3 + 2] * x2 + b[co];
        h[(size_t)co * TLEN + t]    = v;
        h_bf[(size_t)co * TLEN + t] = f2bf(v);
    }
}

// ---------------------------------------------------------------------------
// Core WMMA GEMM body: A pre-swizzled in global (direct b128 fragment loads),
// B staged through 4KB LDS. WG = 256 thr (8 waves), tile 128(M) x 64(N).
// ---------------------------------------------------------------------------
#define GEMM_CORE(Asw, Bm, Kdim)                                                     \
    __shared__ __align__(32) unsigned short sB[32 * 64];                             \
    const int m0 = blockIdx.x * 128;                                                 \
    const int t0 = blockIdx.y * 64;                                                  \
    const int tid = threadIdx.x;                                                     \
    const int wid = tid >> 5, waveM = wid >> 1, waveN = wid & 1;                     \
    const int nkb = (Kdim) >> 5;                                                     \
    v8f acc[2][2] = {};                                                              \
    for (int k0 = 0; k0 < (Kdim); k0 += 32) {                                        \
        stage_B(Bm, sB, k0, t0, 0);                                                  \
        __syncthreads();                                                             \
        const int kb = k0 >> 5;                                                      \
        _Pragma("unroll")                                                            \
        for (int tm = 0; tm < 2; ++tm) {                                             \
            int sub = (m0 >> 4) + waveM * 2 + tm;                                    \
            const unsigned short* ap = (Asw) + ((size_t)sub * nkb + kb) * 512;       \
            if (kb + 1 < nkb) __builtin_prefetch(ap + 512, 0, 0);                    \
            v16bf a = frag_ld(ap);                                                   \
            _Pragma("unroll")                                                        \
            for (int tn = 0; tn < 2; ++tn) {                                         \
                v16bf b = frag_ld(sB + (waveN * 2 + tn) * 512);                      \
                acc[tm][tn] = __builtin_amdgcn_wmma_f32_16x16x32_bf16(               \
                    false, a, false, b, (short)0, acc[tm][tn], false, false);        \
            }                                                                        \
        }                                                                            \
        __syncthreads();                                                             \
    }                                                                                \
    const int lane = tid & 31, half = lane >> 4, lr = lane & 15;

// out[m][t] = A@B + bias[m]
__global__ void k_gemm(const unsigned short* __restrict__ Asw,
                       const unsigned short* __restrict__ Bm, int Kdim,
                       const float* __restrict__ bias, float* __restrict__ out) {
    GEMM_CORE(Asw, Bm, Kdim)
#pragma unroll
    for (int tm = 0; tm < 2; ++tm)
#pragma unroll
        for (int tn = 0; tn < 2; ++tn)
#pragma unroll
            for (int r = 0; r < 8; ++r) {
                int row = m0 + waveM * 32 + tm * 16 + half * 8 + r;
                int col = t0 + waveN * 32 + tn * 16 + lr;
                out[(size_t)row * TLEN + col] = acc[tm][tn][r] + bias[row];
            }
}

// out[m][t] = A@B + bias[m] + fcond[m][t>>9]   (branch-free variant)
__global__ void k_gemm_fcond(const unsigned short* __restrict__ Asw,
                             const unsigned short* __restrict__ Bm, int Kdim,
                             const float* __restrict__ bias, const float* __restrict__ fcond,
                             float* __restrict__ out) {
    GEMM_CORE(Asw, Bm, Kdim)
#pragma unroll
    for (int tm = 0; tm < 2; ++tm)
#pragma unroll
        for (int tn = 0; tn < 2; ++tn)
#pragma unroll
            for (int r = 0; r < 8; ++r) {
                int row = m0 + waveM * 32 + tm * 16 + half * 8 + r;
                int col = t0 + waveN * 32 + tn * 16 + lr;
                out[(size_t)row * TLEN + col] =
                    acc[tm][tn][r] + bias[row] + fcond[row * TEMB + (col >> 9)];
            }
}

// ---------------------------------------------------------------------------
// Dilated conv (3 taps) + cond + gating, fused. Weights pre-swizzled tap-major:
// Wsw[tap][64 subtiles][16 kblocks][512]; filter rows m, gate rows m+512
// (subtile offset +32). Writes pre_bf = sigmoid(f)*tanh(g).
// ---------------------------------------------------------------------------
__global__ void k_dilated(const unsigned short* __restrict__ Wsw,
                          const unsigned short* __restrict__ h_bf, // [512][TLEN]
                          const float* __restrict__ cond,          // [1024][32] incl. biases
                          unsigned short* __restrict__ pre_bf, int dil) {
    __shared__ __align__(32) unsigned short sB[32 * 64];
    const int m0 = blockIdx.x * 128;
    const int t0 = blockIdx.y * 64;
    const int tid = threadIdx.x;
    const int wid = tid >> 5, waveM = wid >> 1, waveN = wid & 1;
    v8f accf[2][2] = {};
    v8f accg[2][2] = {};
    for (int tap = 0; tap < 3; ++tap) {
        const int shift = (2 - tap) * dil;
        const unsigned short* Wt = Wsw + (size_t)tap * 1024 * 512;  // 64 subtiles x 16 kb x 512
        for (int c0 = 0; c0 < WIDTH; c0 += 32) {
            stage_B(h_bf, sB, c0, t0, shift);
            __syncthreads();
            const int kb = c0 >> 5;
#pragma unroll
            for (int tm = 0; tm < 2; ++tm) {
                int sub = (m0 >> 4) + waveM * 2 + tm;
                v16bf af = frag_ld(Wt + ((size_t)sub * 16 + kb) * 512);
                v16bf ag = frag_ld(Wt + ((size_t)(sub + 32) * 16 + kb) * 512);
#pragma unroll
                for (int tn = 0; tn < 2; ++tn) {
                    v16bf b = frag_ld(sB + (waveN * 2 + tn) * 512);
                    accf[tm][tn] = __builtin_amdgcn_wmma_f32_16x16x32_bf16(
                        false, af, false, b, (short)0, accf[tm][tn], false, false);
                    accg[tm][tn] = __builtin_amdgcn_wmma_f32_16x16x32_bf16(
                        false, ag, false, b, (short)0, accg[tm][tn], false, false);
                }
            }
            __syncthreads();
        }
    }
    const int lane = tid & 31, half = lane >> 4, lr = lane & 15;
#pragma unroll
    for (int tm = 0; tm < 2; ++tm)
#pragma unroll
        for (int tn = 0; tn < 2; ++tn)
#pragma unroll
            for (int r = 0; r < 8; ++r) {
                int row = m0 + waveM * 32 + tm * 16 + half * 8 + r;
                int col = t0 + waveN * 32 + tn * 16 + lr;
                int te  = col >> 9;  // nearest upsample of the 32-wide embedding
                float df = accf[tm][tn][r] + cond[row * TEMB + te];
                float dg = accg[tm][tn][r] + cond[(row + 512) * TEMB + te];
                float fgate = 1.f / (1.f + __expf(-df));
                float ggate = tanhf(dg);
                pre_bf[(size_t)row * TLEN + col] = f2bf(fgate * ggate);
            }
}

// ---------------------------------------------------------------------------
// Stacked residual+skip GEMM: Wrs = [w_res(512) ; w_skip(256)] (48 subtiles).
// Block-uniform branch: m0<512 -> h += acc + b_res (and refresh h_bf);
//                       m0>=512 -> skip += acc + b_skip.
// ---------------------------------------------------------------------------
__global__ void k_resskip(const unsigned short* __restrict__ Wrs,
                          const unsigned short* __restrict__ pre_bf,
                          const float* __restrict__ b_res, const float* __restrict__ b_skip,
                          float* __restrict__ h, unsigned short* __restrict__ h_bf,
                          float* __restrict__ skip) {
    GEMM_CORE(Wrs, pre_bf, WIDTH)
    if (m0 < WIDTH) {
#pragma unroll
        for (int tm = 0; tm < 2; ++tm)
#pragma unroll
            for (int tn = 0; tn < 2; ++tn)
#pragma unroll
                for (int r = 0; r < 8; ++r) {
                    int row = m0 + waveM * 32 + tm * 16 + half * 8 + r;
                    int col = t0 + waveN * 32 + tn * 16 + lr;
                    float v = h[(size_t)row * TLEN + col] + acc[tm][tn][r] + b_res[row];
                    h[(size_t)row * TLEN + col]    = v;
                    h_bf[(size_t)row * TLEN + col] = f2bf(v);
                }
    } else {
#pragma unroll
        for (int tm = 0; tm < 2; ++tm)
#pragma unroll
            for (int tn = 0; tn < 2; ++tn)
#pragma unroll
                for (int r = 0; r < 8; ++r) {
                    int sr  = m0 - WIDTH + waveM * 32 + tm * 16 + half * 8 + r;
                    int col = t0 + waveN * 32 + tn * 16 + lr;
                    skip[(size_t)sr * TLEN + col] += acc[tm][tn][r] + b_skip[sr];
                }
    }
}

// ---------------------------------------------------------------------------
extern "C" void kernel_launch(void* const* d_in, const int* in_sizes, int n_in,
                              void* d_out, int out_size, void* d_ws, size_t ws_size,
                              hipStream_t stream) {
    const float* x       = (const float*)d_in[0];
    const float* emb     = (const float*)d_in[1];
    const float* w_init  = (const float*)d_in[2];
    const float* b_init  = (const float*)d_in[3];
    const float* w_iskip = (const float*)d_in[4];
    const float* b_iskip = (const float*)d_in[5];
    const float* w_dil   = (const float*)d_in[6];
    const float* b_dil   = (const float*)d_in[7];
    const float* w_cond  = (const float*)d_in[8];
    const float* b_cond  = (const float*)d_in[9];
    const float* w_res   = (const float*)d_in[10];
    const float* b_res   = (const float*)d_in[11];
    const float* w_skip  = (const float*)d_in[12];
    const float* b_skip  = (const float*)d_in[13];
    const float* w_fskip = (const float*)d_in[14];
    const float* b_fskip = (const float*)d_in[15];
    const float* w_fcond = (const float*)d_in[16];
    const float* b_fcond = (const float*)d_in[17];
    const float* w_quant = (const float*)d_in[18];
    const float* b_quant = (const float*)d_in[19];
    float* out = (float*)d_out;

    uint8_t* p = (uint8_t*)d_ws;
    float* h               = (float*)p;          p += (size_t)WIDTH * TLEN * 4;
    unsigned short* h_bf   = (unsigned short*)p; p += (size_t)WIDTH * TLEN * 2;
    unsigned short* pre_bf = (unsigned short*)p; p += (size_t)WIDTH * TLEN * 2;
    float* skip            = (float*)p;          p += (size_t)SKIPW * TLEN * 4;
    float* cond_all        = (float*)p;          p += (size_t)NSTACK * 1024 * TEMB * 4;
    float* fcond           = (float*)p;          p += (size_t)SKIPW * TEMB * 4;
    unsigned short* wA     = (unsigned short*)p; p += (size_t)3 * 1024 * 512 * 2;  // tap-major swizzled
    unsigned short* wRS    = (unsigned short*)p; p += (size_t)768 * 512 * 2;
    unsigned short* wIS    = (unsigned short*)p; p += (size_t)256 * 512 * 2;
    unsigned short* wFS    = (unsigned short*)p; p += (size_t)256 * 256 * 2;
    unsigned short* wQ     = (unsigned short*)p;

    auto cvt = [&](const float* src, unsigned short* dst, int M, int kbits, int rs, int cs) {
        int n = M << kbits;
        k_cvt_swz<<<(n + 255) / 256, 256, 0, stream>>>(src, dst, M, kbits, rs, cs);
    };

    // Conditioning (folds b_cond + b_dil) and final conditioning
    k_cond<<<(NSTACK * 1024 + 255) / 256, 256, 0, stream>>>(emb, w_cond, b_cond, b_dil, cond_all);
    k_fcond<<<1, 256, 0, stream>>>(emb, w_fcond, b_fcond, fcond);

    // Initial causal conv -> h (f32 + bf16)
    k_init<<<TLEN / 256, 256, 0, stream>>>(x, w_init, b_init, h, h_bf);

    // Static weight conversions (row-major f32 -> swizzled bf16 blocks)
    cvt(w_iskip, wIS, SKIPW, 9, WIDTH, 1);
    cvt(w_fskip, wFS, SKIPW, 8, SKIPW, 1);
    cvt(w_quant, wQ,  QUANTN, 8, SKIPW, 1);

    // skip = w_iskip @ h + b_iskip      (M=256, K=512)
    k_gemm<<<dim3(2, TLEN / 64), 256, 0, stream>>>(wIS, h_bf, WIDTH, b_iskip, skip);

    // 30 sequential residual layers
    for (int i = 0; i < NSTACK; ++i) {
        const int dil = 1 << (i % NLAYERS);
        const float* wd = w_dil + (size_t)i * 1024 * 512 * 3;
        for (int tap = 0; tap < 3; ++tap)                      // de-interleave taps + swizzle
            cvt(wd + tap, wA + (size_t)tap * 1024 * 512, 1024, 9, 512 * 3, 3);
        cvt(w_res + (size_t)i * 512 * 512, wRS, 512, 9, WIDTH, 1);
        cvt(w_skip + (size_t)i * 256 * 512, wRS + (size_t)512 * 512, 256, 9, WIDTH, 1);

        k_dilated<<<dim3(4, TLEN / 64), 256, 0, stream>>>(
            wA, h_bf, cond_all + (size_t)i * 1024 * TEMB, pre_bf, dil);
        k_resskip<<<dim3(6, TLEN / 64), 256, 0, stream>>>(
            wRS, pre_bf, b_res + (size_t)i * 512, b_skip + (size_t)i * 256, h, h_bf, skip);
    }

    // skip = w_fskip @ relu(skip) + b_fskip + fcond_up
    k_relu_to_bf16<<<(SKIPW * TLEN + 255) / 256, 256, 0, stream>>>(skip, pre_bf, SKIPW * TLEN);
    k_gemm_fcond<<<dim3(2, TLEN / 64), 256, 0, stream>>>(wFS, pre_bf, SKIPW, b_fskip, fcond, skip);

    // out = w_quant @ relu(skip) + b_quant
    k_relu_to_bf16<<<(SKIPW * TLEN + 255) / 256, 256, 0, stream>>>(skip, pre_bf, SKIPW * TLEN);
    k_gemm<<<dim3(2, TLEN / 64), 256, 0, stream>>>(wQ, pre_bf, SKIPW, b_quant, out);
}
